// HAMCON_GCN_18107582120776
// MI455X (gfx1250) — matrix operand order repaired and checked by reading, not compiled, not run on hardware
//
#include <hip/hip_runtime.h>
#include <hip/hip_bf16.h>

typedef __bf16 bf16;
typedef __attribute__((ext_vector_type(16))) __bf16 v16bf;
typedef __attribute__((ext_vector_type(8)))  __bf16 v8bf;
typedef __attribute__((ext_vector_type(4)))  __bf16 v4bf;
typedef __attribute__((ext_vector_type(8)))  float  v8f;

#define DD 256
#define DT_STEP 1.0f

// ------------------------------------------------------------------
// Elementwise / graph utility kernels
// ------------------------------------------------------------------
__global__ void k_fill(float* __restrict__ p, float v, long long n) {
  long long i = (long long)blockIdx.x * blockDim.x + threadIdx.x;
  if (i < n) p[i] = v;
}

__global__ void k_deg(const long long* __restrict__ dst, float* __restrict__ deg, long long E) {
  long long e = (long long)blockIdx.x * blockDim.x + threadIdx.x;
  if (e < E) atomicAdd(&deg[dst[e]], 1.0f);
}

__global__ void k_dinv(const float* __restrict__ deg, float* __restrict__ dinv, int N) {
  int i = blockIdx.x * blockDim.x + threadIdx.x;
  if (i < N) {
    float d = deg[i] + 1.0f;           // +1 self loop
    dinv[i] = rsqrtf(fmaxf(d, 1.0f));
  }
}

__global__ void k_t(const long long* __restrict__ src, const long long* __restrict__ dst,
                    const float* __restrict__ dinv, float* __restrict__ t, long long E) {
  long long e = (long long)blockIdx.x * blockDim.x + threadIdx.x;
  if (e < E) atomicAdd(&t[src[e]], dinv[dst[e]]);
}

__global__ void k_s(const float* __restrict__ dinv, const float* __restrict__ t,
                    float* __restrict__ s, int N) {
  int i = blockIdx.x * blockDim.x + threadIdx.x;
  if (i < N) s[i] = dinv[i] * (t[i] + dinv[i]);   // s = A^T 1 (incl self loop)
}

// Convert f32 weight into padded bf16 B-operand, layout (Nout, Kpad) row-major.
// trans==0: dst[n][k] = src[n][k] (src is (srcRows,srcCols))
// trans==1: dst[n][k] = src[k][n]
__global__ void k_convw(bf16* __restrict__ dst, const float* __restrict__ src,
                        int Nout, int Kpad, int srcRows, int srcCols, int trans) {
  long long idx = (long long)blockIdx.x * blockDim.x + threadIdx.x;
  long long total = (long long)Nout * Kpad;
  if (idx >= total) return;
  int n = (int)(idx / Kpad);
  int k = (int)(idx - (long long)n * Kpad);
  float v = 0.0f;
  if (!trans) {
    if (n < srcRows && k < srcCols) v = src[(size_t)n * srcCols + k];
  } else {
    if (k < srcRows && n < srcCols) v = src[(size_t)k * srcCols + n];
  }
  dst[idx] = (bf16)v;
}

// out = (bias?) + dinv^2 * h   (self-loop term of A / A^T, plus bias init)
__global__ void k_init_agg(float* __restrict__ out, const float* __restrict__ h,
                           const float* __restrict__ dinv, const float* __restrict__ bias,
                           int wshift, long long total) {
  long long idx = (long long)blockIdx.x * blockDim.x + threadIdx.x;
  if (idx >= total) return;
  long long i = idx >> wshift;
  int c = (int)(idx & ((1 << wshift) - 1));
  float di = dinv[i];
  float b = bias ? bias[c] : 0.0f;
  out[idx] = b + di * di * h[idx];
}

// Out[to] += dinv[from]*dinv[to] * H[from]  (4 channels per thread)
__global__ void k_scatter(const long long* __restrict__ gfrom, const long long* __restrict__ gto,
                          const float* __restrict__ dinv, const float* __restrict__ H,
                          float* __restrict__ Out, int wshift, long long E) {
  long long gid = (long long)blockIdx.x * blockDim.x + threadIdx.x;
  int cshift = wshift - 2;                 // (width/4) chunks per edge
  long long e = gid >> cshift;
  if (e >= E) return;
  int c = (int)(gid & ((1LL << cshift) - 1)) << 2;
  long long a = gfrom[e], b = gto[e];
  float nrm = dinv[a] * dinv[b];
  const float4 h = *(const float4*)(H + (a << wshift) + c);
  float* o = Out + (b << wshift) + c;
  atomicAdd(o + 0, nrm * h.x);
  atomicAdd(o + 1, nrm * h.y);
  atomicAdd(o + 2, nrm * h.z);
  atomicAdd(o + 3, nrm * h.w);
}

__global__ void k_tanh(float* __restrict__ p, long long n) {
  long long i = (long long)blockIdx.x * blockDim.x + threadIdx.x;
  if (i < n) p[i] = tanhf(p[i]);
}

// g = s[i] * w3[c] * (1 - h2^2)    (width 256)
__global__ void k_gradu2(float* __restrict__ g, const float* __restrict__ h2,
                         const float* __restrict__ s, const float* __restrict__ w3,
                         long long total) {
  long long idx = (long long)blockIdx.x * blockDim.x + threadIdx.x;
  if (idx >= total) return;
  long long i = idx >> 8;
  int c = (int)(idx & 255);
  float h = h2[idx];
  g[idx] = s[i] * w3[c] * (1.0f - h * h);
}

// t *= (1 - h1^2)
__global__ void k_dtanhmul(float* __restrict__ t, const float* __restrict__ h1, long long n) {
  long long i = (long long)blockIdx.x * blockDim.x + threadIdx.x;
  if (i < n) {
    float h = h1[i];
    t[i] *= (1.0f - h * h);
  }
}

// Z[:, :256] += DT*F[:,256:];  Z[:,256:] -= DT*F[:,:256]
__global__ void k_update(float* __restrict__ Z, const float* __restrict__ F, long long total) {
  long long idx = (long long)blockIdx.x * blockDim.x + threadIdx.x;
  if (idx >= total) return;
  long long i = idx >> 8;
  int c = (int)(idx & 255);
  long long base = i << 9;
  Z[base + c]       += DT_STEP * F[base + 256 + c];
  Z[base + 256 + c] -= DT_STEP * F[base + c];
}

// X := Y  (Z[:, :256] = Z[:, 256:])
__global__ void k_copyx(float* __restrict__ Z, long long total) {
  long long idx = (long long)blockIdx.x * blockDim.x + threadIdx.x;
  if (idx >= total) return;
  long long i = idx >> 8;
  int c = (int)(idx & 255);
  long long base = i << 9;
  Z[base + c] = Z[base + 256 + c];
}

// ------------------------------------------------------------------
// WMMA bf16 GEMM:  C(MxN) = A(MxK,f32) * Bt^T  with Bt (N,Kpad) bf16 row-major
// (i.e. C[m,n] = sum_k A[m,k]*Bt[n,k]).  Block tile 128x64, K-step 32, 8 waves.
// Register-prefetch pipeline: global loads of tile k+1 overlap LDS/WMMA of tile k.
// Guards are branch-free (index clamp + select); requires K % 4 == 0.
// mode: 0 = plain store, 1 = +bias, 2 = +bias then relu
// ------------------------------------------------------------------
#define BM 128
#define BN 64
#define BK 32
#define LDS_STRIDE 40   // 80B row pitch: conflict-free 16-row access, 16B aligned

__global__ __launch_bounds__(256)
void k_gemm(const float* __restrict__ A, int lda, int M, int K,
            const bf16* __restrict__ Bt, int Kpad,
            float* __restrict__ C, int ldc, int Nreal,
            const float* __restrict__ bias, int mode) {
  __shared__ bf16 As[BM][LDS_STRIDE];
  __shared__ bf16 Bs[BN][LDS_STRIDE];

  const int tid  = threadIdx.x;
  const int wave = tid >> 5;
  const int lane = tid & 31;
  const int m0 = blockIdx.x * BM;
  const int n0 = blockIdx.y * BN;

  // staging assignments
  const int arow = tid >> 3;          // 0..31 (+ p*32)
  const int akk  = (tid & 7) * 4;     // 0,4,...,28
  const int brow = tid >> 2;          // 0..63
  const int bkk  = (tid & 3) * 8;     // 0,8,16,24

  float4 ra[4];
  v8bf   rb;

  // branch-free fetch of A chunks (index clamped so loads are always legal)
  auto fetchA = [&](int k0) {
    const int gk  = k0 + akk;
    const bool kok = (gk + 4 <= K);          // K % 4 == 0 -> chunk all-in or all-out
    const int gkc = kok ? gk : 0;
    #pragma unroll
    for (int p = 0; p < 4; ++p) {
      const int gm  = m0 + arow + p * 32;
      const bool mok = (gm < M);
      const int gmc = mok ? gm : (M - 1);
      float4 v = *(const float4*)(A + (size_t)gmc * lda + gkc);
      if (!(kok && mok)) { v.x = 0.f; v.y = 0.f; v.z = 0.f; v.w = 0.f; }
      ra[p] = v;
    }
  };
  auto fetchB = [&](int k0) {
    rb = *(const v8bf*)(Bt + (size_t)(n0 + brow) * Kpad + (k0 + bkk));
  };
  auto stage = [&]() {
    #pragma unroll
    for (int p = 0; p < 4; ++p) {
      v4bf c;
      c[0] = (bf16)ra[p].x; c[1] = (bf16)ra[p].y;
      c[2] = (bf16)ra[p].z; c[3] = (bf16)ra[p].w;
      *(v4bf*)&As[arow + p * 32][akk] = c;
    }
    *(v8bf*)&Bs[brow][bkk] = rb;
  };

  v8f acc[4] = {};

  fetchA(0);
  fetchB(0);

  for (int k0 = 0; k0 < Kpad; k0 += BK) {
    stage();
    __syncthreads();

    if (k0 + BK < Kpad) {      // prefetch next tile while computing this one
      fetchA(k0 + BK);
      fetchB(k0 + BK);
    }

    // ---- A fragment: lane L -> row L&15; K chunks (L>>4)*8 and +16 ----
    const int row = lane & 15;
    const int kb  = (lane >> 4) * 8;
    v16bf a;
    {
      const v8bf c0 = *(const v8bf*)&As[wave * 16 + row][kb];
      const v8bf c1 = *(const v8bf*)&As[wave * 16 + row][kb + 16];
      #pragma unroll
      for (int i = 0; i < 8; ++i) { a[i] = c0[i]; a[i + 8] = c1[i]; }
    }
    // ---- B fragments + WMMA: lane L -> col L&15; K chunks (L>>4)*8 and +16 ----
    const int klo = (lane >> 4) * 8;
    #pragma unroll
    for (int nt = 0; nt < 4; ++nt) {
      v16bf b;
      const v8bf d0 = *(const v8bf*)&Bs[nt * 16 + row][klo];
      const v8bf d1 = *(const v8bf*)&Bs[nt * 16 + row][klo + 16];
      #pragma unroll
      for (int i = 0; i < 8; ++i) { b[i] = d0[i]; b[i + 8] = d1[i]; }
      acc[nt] = __builtin_amdgcn_wmma_f32_16x16x32_bf16(
          false, a, false, b, (short)0, acc[nt], false, false);
    }
    __syncthreads();
  }

  // ---- store: lane -> N = lane&15, vgpr r -> M = r + (lane>=16)*8 ----
  const int ncol = lane & 15;
  const int moff = (lane >> 4) * 8;
  const int gmb  = m0 + wave * 16 + moff;
  #pragma unroll
  for (int nt = 0; nt < 4; ++nt) {
    const int gn = n0 + nt * 16 + ncol;
    if (gn >= Nreal) continue;
    const float bv = (mode != 0 && bias) ? bias[gn] : 0.0f;
    #pragma unroll
    for (int r = 0; r < 8; ++r) {
      const int gm = gmb + r;
      if (gm < M) {
        float v = acc[nt][r] + bv;
        if (mode == 2) v = fmaxf(v, 0.0f);
        C[(size_t)gm * ldc + gn] = v;
      }
    }
  }
}

// ------------------------------------------------------------------
// Host-side orchestration
// ------------------------------------------------------------------
static inline dim3 gr1(long long total, int bs) {
  return dim3((unsigned)((total + bs - 1) / bs));
}

extern "C" void kernel_launch(void* const* d_in, const int* in_sizes, int n_in,
                              void* d_out, int out_size, void* d_ws, size_t ws_size,
                              hipStream_t stream) {
  const float*     x     = (const float*)d_in[0];
  const long long* ei    = (const long long*)d_in[1];
  const float*     enc_W = (const float*)d_in[2];
  const float*     enc_b = (const float*)d_in[3];
  const float*     w1    = (const float*)d_in[4];
  const float*     b1    = (const float*)d_in[5];
  const float*     w2    = (const float*)d_in[6];
  const float*     b2    = (const float*)d_in[7];
  const float*     w3    = (const float*)d_in[8];
  const float*     dec_W = (const float*)d_in[10];
  const float*     dec_b = (const float*)d_in[11];

  const int F_IN = in_sizes[2] / DD;            // 500
  const int N    = in_sizes[0] / F_IN;          // 50000
  const long long E = in_sizes[1] / 2;          // 800000
  const int NC   = in_sizes[10] / DD;           // 40
  const long long NL = N;

  const long long* e_src = ei;
  const long long* e_dst = ei + E;

  // ---- carve workspace ----
  char* p = (char*)d_ws;
  auto carve = [&](size_t bytes) -> void* {
    void* r = (void*)p;
    p += (bytes + 255) & ~(size_t)255;
    return r;
  };
  float* Z   = (float*)carve((size_t)N * 512 * 4);  // [X | Y]
  float* H1  = (float*)carve((size_t)N * 512 * 4);
  float* H2  = (float*)carve((size_t)N * 256 * 4);
  float* Ba  = (float*)carve((size_t)N * 512 * 4);
  float* Bb  = (float*)carve((size_t)N * 512 * 4);
  float* Ca  = (float*)carve((size_t)N * 256 * 4);
  float* Cb  = (float*)carve((size_t)N * 256 * 4);
  float* deg = (float*)carve((size_t)N * 4);
  float* tt  = (float*)carve((size_t)N * 4);
  float* dnv = (float*)carve((size_t)N * 4);
  float* sA  = (float*)carve((size_t)N * 4);
  bf16* w1d  = (bf16*)carve((size_t)512 * 512 * 2);
  bf16* w1t  = (bf16*)carve((size_t)512 * 512 * 2);
  bf16* w2d  = (bf16*)carve((size_t)256 * 512 * 2);
  bf16* w2t  = (bf16*)carve((size_t)512 * 256 * 2);
  const int KencP = (F_IN + 31) & ~31;          // 512
  bf16* encd = (bf16*)carve((size_t)256 * KencP * 2);
  bf16* decd = (bf16*)carve((size_t)64 * 256 * 2);

  // ---- graph normalization ----
  k_fill<<<gr1(NL, 256), 256, 0, stream>>>(deg, 0.f, NL);
  k_fill<<<gr1(NL, 256), 256, 0, stream>>>(tt, 0.f, NL);
  k_deg<<<gr1(E, 256), 256, 0, stream>>>(e_dst, deg, E);
  k_dinv<<<gr1(NL, 256), 256, 0, stream>>>(deg, dnv, N);
  k_t<<<gr1(E, 256), 256, 0, stream>>>(e_src, e_dst, dnv, tt, E);
  k_s<<<gr1(NL, 256), 256, 0, stream>>>(dnv, tt, sA, N);

  // ---- weight conversion (padded bf16, both orientations) ----
  k_convw<<<gr1(512LL * 512, 256), 256, 0, stream>>>(w1d, w1, 512, 512, 512, 512, 0);
  k_convw<<<gr1(512LL * 512, 256), 256, 0, stream>>>(w1t, w1, 512, 512, 512, 512, 1);
  k_convw<<<gr1(256LL * 512, 256), 256, 0, stream>>>(w2d, w2, 256, 512, 256, 512, 0);
  k_convw<<<gr1(512LL * 256, 256), 256, 0, stream>>>(w2t, w2, 512, 256, 256, 512, 1);
  k_convw<<<gr1(256LL * KencP, 256), 256, 0, stream>>>(encd, enc_W, 256, KencP, 256, F_IN, 0);
  k_convw<<<gr1(64LL * 256, 256), 256, 0, stream>>>(decd, dec_W, 64, 256, NC, 256, 0);

  auto gemm = [&](const float* A, int lda, int K, const bf16* Bt, int Kpad,
                  float* C, int ldc, int Npad, int Nreal, const float* bias, int mode) {
    dim3 g((unsigned)((N + BM - 1) / BM), (unsigned)(Npad / BN));
    k_gemm<<<g, 256, 0, stream>>>(A, lda, N, K, Bt, Kpad, C, ldc, Nreal, bias, mode);
  };

  // ---- encoder: Y = relu(x @ enc_W^T + enc_b) into Z[:,256:], then X = Y ----
  gemm(x, F_IN, F_IN, encd, KencP, Z + 256, 512, 256, 256, enc_b, 2);
  k_copyx<<<gr1(NL * 256, 256), 256, 0, stream>>>(Z, NL * 256);

  // ---- 3 Hamiltonian flow steps ----
  for (int it = 0; it < 3; ++it) {
    // forward layer 1: u1 = A(Z w1^T) + b1; h1 = tanh(u1)
    gemm(Z, 512, 512, w1d, 512, Ba, 512, 512, 512, nullptr, 0);
    k_init_agg<<<gr1(NL * 512, 256), 256, 0, stream>>>(H1, Ba, dnv, b1, 9, NL * 512);
    k_scatter<<<gr1(E << 7, 256), 256, 0, stream>>>(e_src, e_dst, dnv, Ba, H1, 9, E);
    k_tanh<<<gr1(NL * 512, 256), 256, 0, stream>>>(H1, NL * 512);
    // forward layer 2: u2 = A(h1 w2^T) + b2; h2 = tanh(u2)
    gemm(H1, 512, 512, w2d, 512, Ca, 256, 256, 256, nullptr, 0);
    k_init_agg<<<gr1(NL * 256, 256), 256, 0, stream>>>(H2, Ca, dnv, b2, 8, NL * 256);
    k_scatter<<<gr1(E << 6, 256), 256, 0, stream>>>(e_src, e_dst, dnv, Ca, H2, 8, E);
    k_tanh<<<gr1(NL * 256, 256), 256, 0, stream>>>(H2, NL * 256);
    // backward: g2 = s*w3*(1-h2^2)
    k_gradu2<<<gr1(NL * 256, 256), 256, 0, stream>>>(Ca, H2, sA, w3, NL * 256);
    // AG2 = A^T g2
    k_init_agg<<<gr1(NL * 256, 256), 256, 0, stream>>>(Cb, Ca, dnv, nullptr, 8, NL * 256);
    k_scatter<<<gr1(E << 6, 256), 256, 0, stream>>>(e_dst, e_src, dnv, Ca, Cb, 8, E);
    // g1 = (AG2 @ w2) * (1 - h1^2)
    gemm(Cb, 256, 256, w2t, 256, Ba, 512, 512, 512, nullptr, 0);
    k_dtanhmul<<<gr1(NL * 512, 256), 256, 0, stream>>>(Ba, H1, NL * 512);
    // AG1 = A^T g1
    k_init_agg<<<gr1(NL * 512, 256), 256, 0, stream>>>(Bb, Ba, dnv, nullptr, 9, NL * 512);
    k_scatter<<<gr1(E << 7, 256), 256, 0, stream>>>(e_dst, e_src, dnv, Ba, Bb, 9, E);
    // f = AG1 @ w1
    gemm(Bb, 512, 512, w1t, 512, Ba, 512, 512, 512, nullptr, 0);
    // leapfrog update of [X|Y]
    k_update<<<gr1(NL * 256, 256), 256, 0, stream>>>(Z, Ba, NL * 256);
  }

  // ---- decoder: out = X @ dec_W^T + dec_b ----
  gemm(Z, 512, 256, decd, 256, (float*)d_out, NC, 64, NC, dec_b, 1);
}
